// Attention_14817637171563
// MI455X (gfx1250) — matrix operand order repaired and checked by reading, not compiled, not run on hardware
//
#include <hip/hip_runtime.h>
#include <hip/hip_bf16.h>

// ---------------------------------------------------------------------------
// Types for CDNA5 (gfx1250) WMMA: wave32, 16x16x32 bf16 -> f32
// ---------------------------------------------------------------------------
typedef __attribute__((ext_vector_type(16))) __bf16 v16bf;
typedef __attribute__((ext_vector_type(8)))  float  v8f;

struct alignas(16) U128 { unsigned int w[4]; };

union FragBF {
    v16bf v;
    unsigned short u[16];
    U128 q[2];
};

static constexpr int Lq = 1536;   // sequence length
static constexpr int Cc = 768;    // channels
static constexpr int Hh = 8;      // heads
static constexpr int Dk = 64;     // head dim (qk)
static constexpr int Dv = 96;     // head dim (v)
static constexpr int Ff = 96;     // pos features
static constexpr int P2 = 3072;   // padded 2L-1 rows for pos_k (3071 + 1 pad)

__device__ __forceinline__ unsigned short f2bf(float f) {
    unsigned int u = __float_as_uint(f);
    unsigned int r = (u + 0x7FFFu + ((u >> 16) & 1u)) >> 16;
    return (unsigned short)r;
}

__device__ __forceinline__ v8f vzero8() {
    v8f x;
    for (int i = 0; i < 8; ++i) x[i] = 0.0f;
    return x;
}

// A fragment: 16x32 bf16 tile of row-major A (lda in elements), tile at (row0,k0).
// ISA layout: lane<16 holds row=lane, K {0..7,16..23}; lane>=16 holds K {8..15,24..31}.
__device__ __forceinline__ v16bf load_a16x32(const unsigned short* A, int lda,
                                             int row0, int k0) {
    int lane = threadIdx.x & 31;
    int r = lane & 15;
    int kb = k0 + ((lane >> 4) << 3);
    const unsigned short* p = A + (size_t)(row0 + r) * lda + kb;
    FragBF f;
    f.q[0] = *(const U128*)(p);
    f.q[1] = *(const U128*)(p + 16);
    return f.v;
}

// B fragment: 32x16 (KxN) from N-major storage Bt[n][k] (ldb in elements).
// ISA layout: lane<16 holds col=lane&15, K {0..7,16..23}; lane>=16 holds K {8..15,24..31}.
__device__ __forceinline__ v16bf load_b32x16(const unsigned short* Bt, int ldb,
                                             int n0, int k0) {
    int lane = threadIdx.x & 31;
    int n = lane & 15;
    int kb = k0 + ((lane >> 4) << 3);
    const unsigned short* p = Bt + (size_t)(n0 + n) * ldb + kb;
    FragBF f;
    f.q[0] = *(const U128*)(p);
    f.q[1] = *(const U128*)(p + 16);
    return f.v;
}

__device__ __forceinline__ v8f wmma_bf16(v16bf a, v16bf b, v8f c) {
    return __builtin_amdgcn_wmma_f32_16x16x32_bf16(false, a, false, b,
                                                   (short)0, c, false, false);
}

// ---------------------------------------------------------------------------
// Elementwise converts
// ---------------------------------------------------------------------------
__global__ void k_cvt_bf16(const float* __restrict__ in,
                           unsigned short* __restrict__ out, int n) {
    int i = blockIdx.x * 256 + threadIdx.x;
    if (i < n) out[i] = f2bf(in[i]);
}

// W (K x N, row-major) -> WT (N x K) bf16
__global__ void k_transpose_bf16(const float* __restrict__ W,
                                 unsigned short* __restrict__ WT, int K, int N) {
    int idx = blockIdx.x * 256 + threadIdx.x;
    if (idx < K * N) {
        int k = idx / N, n = idx % N;
        WT[(size_t)n * K + k] = f2bf(W[idx]);
    }
}

// ---------------------------------------------------------------------------
// bf16 WMMA GEMM: C[M,N] f32 = A[M,K]bf16 @ B (stored N-major Bt[N,K]) + bias
// block = 128 threads (4 waves in 2x2), block tile 64x64, wave tile 32x32.
// ---------------------------------------------------------------------------
__global__ void k_gemm_bf16(const unsigned short* __restrict__ A,
                            const unsigned short* __restrict__ Bt,
                            float* __restrict__ C, const float* __restrict__ bias,
                            int M, int N, int K) {
    int wave = threadIdx.x >> 5;
    int lane = threadIdx.x & 31;
    int m0 = blockIdx.y * 64 + (wave >> 1) * 32;
    int n0 = blockIdx.x * 64 + (wave & 1) * 32;

    v8f c00 = vzero8(), c01 = vzero8(), c10 = vzero8(), c11 = vzero8();

    for (int k0 = 0; k0 < K; k0 += 32) {
        v16bf a0 = load_a16x32(A, K, m0,      k0);
        v16bf a1 = load_a16x32(A, K, m0 + 16, k0);
        v16bf b0 = load_b32x16(Bt, K, n0,      k0);
        v16bf b1 = load_b32x16(Bt, K, n0 + 16, k0);
        c00 = wmma_bf16(a0, b0, c00);
        c01 = wmma_bf16(a0, b1, c01);
        c10 = wmma_bf16(a1, b0, c10);
        c11 = wmma_bf16(a1, b1, c11);
    }

    int half = lane >> 4, col = lane & 15;
    float b0v = bias ? bias[n0 + col] : 0.0f;
    float b1v = bias ? bias[n0 + 16 + col] : 0.0f;
    for (int e = 0; e < 8; ++e) {
        int r0 = m0 + e + 8 * half;
        int r1 = r0 + 16;
        C[(size_t)r0 * N + n0 + col]       = c00[e] + b0v;
        C[(size_t)r0 * N + n0 + 16 + col]  = c01[e] + b1v;
        C[(size_t)r1 * N + n0 + col]       = c10[e] + b0v;
        C[(size_t)r1 * N + n0 + 16 + col]  = c11[e] + b1v;
    }
}

// ---------------------------------------------------------------------------
// Head prep: qc = q/8 + content_bias, qp = q/8 + pos_bias, k -> bf16 [H][L][64]
// ---------------------------------------------------------------------------
__global__ void k_prep_qk(const float* __restrict__ qf, const float* __restrict__ kf,
                          const float* __restrict__ cb, const float* __restrict__ pb,
                          unsigned short* __restrict__ qc,
                          unsigned short* __restrict__ qp,
                          unsigned short* __restrict__ km) {
    int idx = blockIdx.x * 256 + threadIdx.x;  // over L * H*Dk
    if (idx >= Lq * Hh * Dk) return;
    int i = idx / (Hh * Dk), c = idx % (Hh * Dk);
    int h = c >> 6, d = c & 63;
    size_t o = ((size_t)h * Lq + i) * Dk + d;
    float q = qf[idx] * 0.125f;  // 1/sqrt(DK)
    qc[o] = f2bf(q + cb[c]);
    qp[o] = f2bf(q + pb[c]);
    km[o] = f2bf(kf[idx]);
}

// v [L][H*Dv] f32 -> vT bf16 [H][Dv][L]
__global__ void k_prep_v(const float* __restrict__ vf,
                         unsigned short* __restrict__ vT) {
    int idx = blockIdx.x * 256 + threadIdx.x;  // over L * H*Dv
    if (idx >= Lq * Hh * Dv) return;
    int i = idx / (Hh * Dv), c = idx % (Hh * Dv);
    int h = c / Dv, d = c % Dv;
    vT[((size_t)h * Dv + d) * Lq + i] = f2bf(vf[idx]);
}

// ---------------------------------------------------------------------------
// pos_k via suffix sums of Wpos (mask is monotone in feature index):
// S[f][c] = sum_{g>=f} Wpos[g][c];  pos_k[j][c] = S[floor(log2(|j-(L-1)|+1))][c]
// ---------------------------------------------------------------------------
__global__ void k_wpos_suffix(const float* __restrict__ Wpos,
                              float* __restrict__ S) {
    int c = blockIdx.x * 256 + threadIdx.x;
    if (c >= Hh * Dk) return;
    float acc = 0.0f;
    for (int f = Ff - 1; f >= 0; --f) {
        acc += Wpos[f * (Hh * Dk) + c];
        S[f * (Hh * Dk) + c] = acc;
    }
}

__global__ void k_build_posk(const float* __restrict__ S,
                             unsigned short* __restrict__ posk) {
    int idx = blockIdx.x * 256 + threadIdx.x;  // over P2 * H*Dk
    if (idx >= P2 * Hh * Dk) return;
    int j = idx / (Hh * Dk), c = idx % (Hh * Dk);
    if (j >= 2 * Lq - 1) { posk[idx] = 0; return; }  // pad row
    int p = j - (Lq - 1);
    int ap = p < 0 ? -p : p;
    int f = 31 - __clz(ap + 1);   // smallest f with 2^(f+1)-1 > |p|
    posk[idx] = f2bf(S[f * (Hh * Dk) + c]);
}

// ---------------------------------------------------------------------------
// Flash attention with relative-position logits.
// grid = (L/64, H), block = 128 (4 waves); wave owns 16 query rows, scans keys
// in blocks of 32. The pos-window R for consecutive 16-key subtiles overlaps
// by 16 rows, so the low half is carried in registers across subtiles:
// steady state per 16-key subtile = 2 content WMMA + 2 pos WMMA, plus 6
// P@V WMMA per 32-key block. Toeplitz gather P[i,j] += R[i, j-i+15] via LDS.
// ---------------------------------------------------------------------------
__global__ void k_flash(const unsigned short* __restrict__ qc,
                        const unsigned short* __restrict__ qp,
                        const unsigned short* __restrict__ kmat,
                        const unsigned short* __restrict__ vT,
                        const unsigned short* __restrict__ posk,
                        unsigned short* __restrict__ attn_out) {
    __shared__ float          Rlds[4][16][32];
    __shared__ unsigned short Plds[4][16][32];

    int wave = threadIdx.x >> 5;
    int lane = threadIdx.x & 31;
    int h = blockIdx.y;
    int i0 = blockIdx.x * 64 + wave * 16;

    const unsigned short* qch = qc   + (size_t)h * Lq * Dk;
    const unsigned short* qph = qp   + (size_t)h * Lq * Dk;
    const unsigned short* kh  = kmat + (size_t)h * Lq * Dk;
    const unsigned short* vh  = vT   + (size_t)h * Dv * Lq;
    const unsigned short* pk  = posk + h * Dk;   // ldb = H*Dk = 512

    // Loop-invariant A fragments (K = 64 = two 32-chunks)
    v16bf aqc0 = load_a16x32(qch, Dk, i0, 0);
    v16bf aqc1 = load_a16x32(qch, Dk, i0, 32);
    v16bf aqp0 = load_a16x32(qph, Dk, i0, 0);
    v16bf aqp1 = load_a16x32(qph, Dk, i0, 32);

    v8f o[6];
    for (int n = 0; n < 6; ++n) o[n] = vzero8();
    float m[8], lsum[8];
    for (int e = 0; e < 8; ++e) { m[e] = -3.0e38f; lsum[e] = 0.0f; }

    int half = lane >> 4, col = lane & 15;

    // Prime the low pos-window: R rows (L-1)-15-i0 .. +15 (t' = -15..0).
    v8f rlo = vzero8();
    {
        int jrow_lo = (Lq - 1) - 15 - i0;   // jt = 0
        rlo = wmma_bf16(aqp0, load_b32x16(pk, Hh * Dk, jrow_lo, 0),  rlo);
        rlo = wmma_bf16(aqp1, load_b32x16(pk, Hh * Dk, jrow_lo, 32), rlo);
    }

    for (int jb = 0; jb < Lq; jb += 32) {
        v8f s[2];
        for (int t = 0; t < 2; ++t) {
            int jt = jb + 16 * t;
            // content logits: qc . k^T  (k natural layout == B fragment)
            v8f acc = vzero8();
            acc = wmma_bf16(aqc0, load_b32x16(kh, Dk, jt, 0),  acc);
            acc = wmma_bf16(aqc1, load_b32x16(kh, Dk, jt, 32), acc);

            // pos window high half (rows delta+1 .. delta+16); low half is rlo
            int jrow0 = jt - i0 + (Lq - 1) - 15;   // in [0, 3040]
            v8f rhi = vzero8();
            rhi = wmma_bf16(aqp0, load_b32x16(pk, Hh * Dk, jrow0 + 16, 0),  rhi);
            rhi = wmma_bf16(aqp1, load_b32x16(pk, Hh * Dk, jrow0 + 16, 32), rhi);

            // spill R to LDS (C layout: row = e+8*half, col = lane&15)
            for (int e = 0; e < 8; ++e) {
                Rlds[wave][e + 8 * half][col]      = rlo[e];
                Rlds[wave][e + 8 * half][col + 16] = rhi[e];
            }
            // Toeplitz gather: P[i,j] += R[i, (j - i) + 15]
            for (int e = 0; e < 8; ++e) {
                int r = e + 8 * half;
                acc[e] += Rlds[wave][r][col - r + 15];
            }
            s[t] = acc;
            rlo = rhi;   // window slides by 16 for the next subtile
        }

        // online softmax over the 32-key block (rows striped per half-wave)
        float scale[8];
        for (int e = 0; e < 8; ++e) {
            float mx = fmaxf(s[0][e], s[1][e]);
            for (int msk = 1; msk < 16; msk <<= 1)
                mx = fmaxf(mx, __shfl_xor(mx, msk, 32));
            float mn = fmaxf(m[e], mx);
            scale[e] = __expf(m[e] - mn);
            m[e] = mn;
            float p0 = __expf(s[0][e] - mn);
            float p1 = __expf(s[1][e] - mn);
            float rs = p0 + p1;
            for (int msk = 1; msk < 16; msk <<= 1)
                rs += __shfl_xor(rs, msk, 32);
            lsum[e] = lsum[e] * scale[e] + rs;
            int r = e + 8 * half;
            Plds[wave][r][col]      = f2bf(p0);
            Plds[wave][r][col + 16] = f2bf(p1);
        }
        for (int n = 0; n < 6; ++n)
            for (int e = 0; e < 8; ++e) o[n][e] *= scale[e];

        // repack P (16x32) from LDS into A-fragment layout
        FragBF fp;
        {
            int r = lane & 15;
            int kb = (lane >> 4) << 3;
            const unsigned short* p = &Plds[wave][r][kb];
            for (int e2 = 0; e2 < 8; ++e2) {
                fp.u[e2]     = p[e2];
                fp.u[8 + e2] = p[16 + e2];
            }
        }
        // o += P @ V  (vT[n][k] natural B layout, K = 32 keys)
        for (int n = 0; n < 6; ++n)
            o[n] = wmma_bf16(fp.v, load_b32x16(vh, Lq, n * 16, jb), o[n]);
    }

    // finalize: divide by softmax denom, store bf16 [L][H*Dv]
    for (int e = 0; e < 8; ++e) {
        float inv = 1.0f / lsum[e];
        int r = i0 + e + 8 * half;
        for (int n = 0; n < 6; ++n)
            attn_out[(size_t)r * (Hh * Dv) + h * Dv + n * 16 + col] =
                f2bf(o[n][e] * inv);
    }
}

// ---------------------------------------------------------------------------
// Host-side launch
// ---------------------------------------------------------------------------
extern "C" void kernel_launch(void* const* d_in, const int* in_sizes, int n_in,
                              void* d_out, int out_size, void* d_ws, size_t ws_size,
                              hipStream_t stream) {
    const float* x    = (const float*)d_in[0];
    const float* Wq   = (const float*)d_in[1];
    const float* Wk   = (const float*)d_in[2];
    const float* Wv   = (const float*)d_in[3];
    const float* Wout = (const float*)d_in[4];
    const float* bout = (const float*)d_in[5];
    const float* Wpos = (const float*)d_in[6];
    const float* cb   = (const float*)d_in[7];
    const float* pb   = (const float*)d_in[8];
    float* out        = (float*)d_out;

    unsigned char* ws = (unsigned char*)d_ws;
    size_t off = 0;
    auto alloc = [&](size_t bytes) -> void* {
        void* p = ws + off;
        off = (off + bytes + 255) & ~(size_t)255;
        return p;
    };

    unsigned short* xb    = (unsigned short*)alloc((size_t)Lq * Cc * 2);
    unsigned short* WqT   = (unsigned short*)alloc((size_t)Cc * Hh * Dk * 2);
    unsigned short* WkT   = (unsigned short*)alloc((size_t)Cc * Hh * Dk * 2);
    unsigned short* WvT   = (unsigned short*)alloc((size_t)Cc * Hh * Dv * 2);
    unsigned short* WoutT = (unsigned short*)alloc((size_t)Hh * Dv * Cc * 2);
    float* qf             = (float*)alloc((size_t)Lq * Hh * Dk * 4);
    float* kf             = (float*)alloc((size_t)Lq * Hh * Dk * 4);
    float* vf             = (float*)alloc((size_t)Lq * Hh * Dv * 4);
    unsigned short* qcb   = (unsigned short*)alloc((size_t)Hh * Lq * Dk * 2);
    unsigned short* qpb   = (unsigned short*)alloc((size_t)Hh * Lq * Dk * 2);
    unsigned short* kmb   = (unsigned short*)alloc((size_t)Hh * Lq * Dk * 2);
    unsigned short* vTb   = (unsigned short*)alloc((size_t)Hh * Dv * Lq * 2);
    // region reuse (qf/kf/vf are dead after k_prep_*):
    float* S              = (float*)vf;                 // 96*512*4   = 192 KB
    unsigned short* posk  = (unsigned short*)qf;        // 3072*512*2 = 3 MB (== qf size)
    unsigned short* attnb = (unsigned short*)kf;        // 1536*768*2 = 2.25 MB (< kf size)

    auto blocks = [](int n) { return (n + 255) / 256; };

    // 1) converts / transposes
    k_cvt_bf16<<<blocks(Lq * Cc), 256, 0, stream>>>(x, xb, Lq * Cc);
    k_transpose_bf16<<<blocks(Cc * Hh * Dk), 256, 0, stream>>>(Wq, WqT, Cc, Hh * Dk);
    k_transpose_bf16<<<blocks(Cc * Hh * Dk), 256, 0, stream>>>(Wk, WkT, Cc, Hh * Dk);
    k_transpose_bf16<<<blocks(Cc * Hh * Dv), 256, 0, stream>>>(Wv, WvT, Cc, Hh * Dv);
    k_transpose_bf16<<<blocks(Hh * Dv * Cc), 256, 0, stream>>>(Wout, WoutT, Hh * Dv, Cc);

    // 2) QKV projections (WMMA GEMM)
    k_gemm_bf16<<<dim3((Hh * Dk) / 64, Lq / 64), 128, 0, stream>>>(
        xb, WqT, qf, nullptr, Lq, Hh * Dk, Cc);
    k_gemm_bf16<<<dim3((Hh * Dk) / 64, Lq / 64), 128, 0, stream>>>(
        xb, WkT, kf, nullptr, Lq, Hh * Dk, Cc);
    k_gemm_bf16<<<dim3((Hh * Dv) / 64, Lq / 64), 128, 0, stream>>>(
        xb, WvT, vf, nullptr, Lq, Hh * Dv, Cc);

    // 3) head prep (consumes qf/kf/vf)
    k_prep_qk<<<blocks(Lq * Hh * Dk), 256, 0, stream>>>(qf, kf, cb, pb, qcb, qpb, kmb);
    k_prep_v<<<blocks(Lq * Hh * Dv), 256, 0, stream>>>(vf, vTb);

    // 4) relative-position table (reuses vf region for S, qf region for posk)
    k_wpos_suffix<<<blocks(Hh * Dk), 256, 0, stream>>>(Wpos, S);
    k_build_posk<<<blocks(P2 * Hh * Dk), 256, 0, stream>>>(S, posk);

    // 5) flash attention with relative shift
    k_flash<<<dim3(Lq / 64, Hh), 128, 0, stream>>>(qcb, qpb, kmb, vTb, posk, attnb);

    // 6) output projection with bias -> f32 d_out
    k_gemm_bf16<<<dim3(Cc / 64, Lq / 64), 128, 0, stream>>>(
        attnb, WoutT, out, bout, Lq, Cc, Hh * Dv);
}
